// GatedSlotAttention_75969381531971
// MI455X (gfx1250) — matrix operand order repaired and checked by reading, not compiled, not run on hardware
//
#include <hip/hip_runtime.h>
#include <stdint.h>

// ---------------------------------------------------------------------------
// Types
// ---------------------------------------------------------------------------
typedef __attribute__((ext_vector_type(16))) __bf16 v16bf;
typedef __attribute__((ext_vector_type(8)))  float  v8f;
typedef unsigned short u16;
typedef __attribute__((ext_vector_type(8)))  u16    v8us;

#define T_SEQ   2048
#define BATCH   4
#define DMODEL  2048
#define NHEAD   8
#define DKH     256
#define MSLOT   64
#define YCOLS   6656   // 3*2048 (q,k,v) + 512 (f)
#define KDIM    2048
#define LDSPITCH 40    // 32 K-elems + 8 pad (80B row stride -> conflict-free)

__device__ __forceinline__ u16 f2bf(float f) {
    __bf16 b = (__bf16)f;
    return __builtin_bit_cast(u16, b);
}
__device__ __forceinline__ float bf2f(u16 u) {
    __bf16 b = __builtin_bit_cast(__bf16, u);
    return (float)b;
}
__device__ __forceinline__ float swishf(float x) { return x / (1.f + __expf(-x)); }
__device__ __forceinline__ float logsigf(float x) {
    return (x < 0.f) ? (x - log1pf(__expf(x))) : (-log1pf(__expf(-x)));
}

// ---------------------------------------------------------------------------
// fp32 -> bf16 weight conversion (4 elems / thread)
// ---------------------------------------------------------------------------
__global__ __launch_bounds__(256) void cvt_kernel(const float* __restrict__ in,
                                                  u16* __restrict__ out, int n) {
    int i = (blockIdx.x * 256 + threadIdx.x) * 4;
    if (i + 3 < n) {
        float4 v = *(const float4*)(in + i);
        out[i + 0] = f2bf(v.x);
        out[i + 1] = f2bf(v.y);
        out[i + 2] = f2bf(v.z);
        out[i + 3] = f2bf(v.w);
    }
}

// ---------------------------------------------------------------------------
// RMSNorm (optionally preceded by swish) + cast to bf16.  One row per block.
// ---------------------------------------------------------------------------
__global__ __launch_bounds__(256) void rmsnorm_cast_kernel(
    const float* __restrict__ x, const float* __restrict__ w,
    u16* __restrict__ out, int do_swish) {
    const int row  = blockIdx.x;
    const int tid  = threadIdx.x;
    const int lane = tid & 31;
    const int wid  = tid >> 5;

    const float4* xr = (const float4*)(x + (size_t)row * DMODEL);
    float4 a = xr[tid];
    float4 b = xr[tid + 256];
    if (do_swish) {
        a.x = swishf(a.x); a.y = swishf(a.y); a.z = swishf(a.z); a.w = swishf(a.w);
        b.x = swishf(b.x); b.y = swishf(b.y); b.z = swishf(b.z); b.w = swishf(b.w);
    }
    float ss = a.x*a.x + a.y*a.y + a.z*a.z + a.w*a.w
             + b.x*b.x + b.y*b.y + b.z*b.z + b.w*b.w;
    #pragma unroll
    for (int off = 16; off; off >>= 1) ss += __shfl_xor(ss, off, 32);
    __shared__ float sred[8];
    if (lane == 0) sred[wid] = ss;
    __syncthreads();
    float tot = 0.f;
    #pragma unroll
    for (int i = 0; i < 8; ++i) tot += sred[i];
    const float rstd = rsqrtf(tot * (1.f / (float)DMODEL) + 1e-5f);

    const float4* wr = (const float4*)w;
    float4 w0 = wr[tid], w1 = wr[tid + 256];
    u16* orow = out + (size_t)row * DMODEL;
    int c0 = tid * 4, c1 = (tid + 256) * 4;
    orow[c0 + 0] = f2bf(a.x * rstd * w0.x);
    orow[c0 + 1] = f2bf(a.y * rstd * w0.y);
    orow[c0 + 2] = f2bf(a.z * rstd * w0.z);
    orow[c0 + 3] = f2bf(a.w * rstd * w0.w);
    orow[c1 + 0] = f2bf(b.x * rstd * w1.x);
    orow[c1 + 1] = f2bf(b.y * rstd * w1.y);
    orow[c1 + 2] = f2bf(b.z * rstd * w1.z);
    orow[c1 + 3] = f2bf(b.w * rstd * w1.w);
}

// ---------------------------------------------------------------------------
// Async staging of one K-step tile pair into LDS (bf16), padded pitch.
//   A tile: 128 rows x 32 K  (512 x 16B chunks, 2/thread)
//   B tile: 256 rows x 32 K  (1024 x 16B chunks, 4/thread)
// ---------------------------------------------------------------------------
__device__ __forceinline__ void stage_tiles(const u16* A, const u16* W,
                                            u16* la, u16* lb, int tid,
                                            int blockM, int blockN, int K, int kk) {
    #pragma unroll
    for (int t = 0; t < 2; ++t) {
        int c   = tid + t * 256;
        int row = c >> 2;
        int ko  = (c & 3) * 8;
        unsigned goff  = (unsigned)(((size_t)(blockM + row) * K + kk + ko) * 2);
        unsigned laddr = (unsigned)(uintptr_t)(la + row * LDSPITCH + ko);
        asm volatile("global_load_async_to_lds_b128 %0, %1, %2 offset:0"
                     :: "v"(laddr), "v"(goff), "s"(A) : "memory");
    }
    #pragma unroll
    for (int t = 0; t < 4; ++t) {
        int c   = tid + t * 256;
        int row = c >> 2;
        int ko  = (c & 3) * 8;
        unsigned goff  = (unsigned)(((size_t)(blockN + row) * K + kk + ko) * 2);
        unsigned laddr = (unsigned)(uintptr_t)(lb + row * LDSPITCH + ko);
        asm volatile("global_load_async_to_lds_b128 %0, %1, %2 offset:0"
                     :: "v"(laddr), "v"(goff), "s"(W) : "memory");
    }
}

// ---------------------------------------------------------------------------
// WMMA GEMM: C[token, ocol] = A_bf16[8192, K] x W_bf16[Ncols, K]^T
// Block = 128(M) x 256(N), 8 waves as 2(M) x 4(N); wave tile 64x64 =
// 4x4 WMMA 16x16x32 tiles (16 v_wmma per K-step).  LDS double-buffered,
// filled by CDNA5 async-to-LDS DMA (ASYNCcnt), one barrier per K-step.
// mode 0: fp32 store.  mode 1: bf16 store, swish (col<6144) / logsig/8.
// ---------------------------------------------------------------------------
__global__ __launch_bounds__(256) void wmma_gemm_kernel(
    const u16* __restrict__ A, const u16* __restrict__ W,
    void* __restrict__ out, int Ncols, int K, int mode) {
    const int tid  = threadIdx.x;
    const int lane = tid & 31;
    const int wv   = tid >> 5;     // 0..7
    const int wm   = wv & 1;       // 2 row-waves
    const int wn   = wv >> 1;      // 4 col-waves
    const int blockM = blockIdx.x * 128;
    const int blockN = blockIdx.y * 256;

    __shared__ alignas(16) u16 lds_a[2][128 * LDSPITCH];
    __shared__ alignas(16) u16 lds_b[2][256 * LDSPITCH];

    v8f acc[4][4] = {};
    union Frag { v16bf v; v8us h[2]; };

    const int nk = K / 32;
    stage_tiles(A, W, lds_a[0], lds_b[0], tid, blockM, blockN, K, 0);

    for (int ki = 0; ki < nk; ++ki) {
        asm volatile("s_wait_asynccnt 0" ::: "memory");
        __syncthreads();
        if (ki + 1 < nk)
            stage_tiles(A, W, lds_a[(ki + 1) & 1], lds_b[(ki + 1) & 1], tid,
                        blockM, blockN, K, (ki + 1) * 32);

        const u16* la = lds_a[ki & 1];
        const u16* lb = lds_b[ki & 1];

        // fragments (ISA 7.12.2 layouts)
        Frag af[4], bfr[4];
        const int akb = (lane < 16) ? 0 : 8;   // A: K {0..7,16..23}|{8..15,24..31}
        const int bkb = (lane < 16) ? 0 : 8;   // B: K {0..7,16..23}|{8..15,24..31}
        #pragma unroll
        for (int r = 0; r < 4; ++r) {
            int lrow = wm * 64 + r * 16 + (lane & 15);
            af[r].h[0] = *(const v8us*)&la[lrow * LDSPITCH + akb];
            af[r].h[1] = *(const v8us*)&la[lrow * LDSPITCH + akb + 16];
        }
        #pragma unroll
        for (int c = 0; c < 4; ++c) {
            int brow = wn * 64 + c * 16 + (lane & 15);
            bfr[c].h[0] = *(const v8us*)&lb[brow * LDSPITCH + bkb];
            bfr[c].h[1] = *(const v8us*)&lb[brow * LDSPITCH + bkb + 16];
        }
        #pragma unroll
        for (int r = 0; r < 4; ++r)
            #pragma unroll
            for (int c = 0; c < 4; ++c)
                acc[r][c] = __builtin_amdgcn_wmma_f32_16x16x32_bf16(
                    false, af[r].v, false, bfr[c].v, (short)0, acc[r][c],
                    false, false);
    }

    // ---- epilogue: C layout = VGPR i, lane l -> row i+8*(l>>4), col l&15 ----
    #pragma unroll
    for (int r = 0; r < 4; ++r) {
        #pragma unroll
        for (int c = 0; c < 4; ++c) {
            #pragma unroll
            for (int i = 0; i < 8; ++i) {
                int grow = blockM + wm * 64 + r * 16 + i + 8 * (lane >> 4);
                int gcol = blockN + wn * 64 + c * 16 + (lane & 15);
                float v = acc[r][c][i];
                if (mode == 1) {
                    float a = (gcol < 3 * DMODEL) ? swishf(v) : logsigf(v) * 0.125f;
                    ((u16*)out)[(size_t)grow * Ncols + gcol] = f2bf(a);
                } else {
                    ((float*)out)[(size_t)grow * Ncols + gcol] = v;
                }
            }
        }
    }
}

// ---------------------------------------------------------------------------
// Sequential GSA scan.  One 512-thread workgroup per (b,h) chain (32 total).
// State register-resident:
//   thread t = 64*c + m (c=0..7):  Hk[d][m] for d in [32c, 32c+32)   (32 regs)
//   thread t = 256*r + dv (r=0..1): Hv[m][dv] for m in [32r, 32r+32) (32 regs)
// ---------------------------------------------------------------------------
__global__ __launch_bounds__(512) void gsa_scan_kernel(
    const u16* __restrict__ Y, float* __restrict__ O) {
    const int bh  = blockIdx.x;
    const int b   = bh >> 3;
    const int h   = bh & 7;
    const int tid = threadIdx.x;

    const int cK = tid >> 6;   // 0..7
    const int mK = tid & 63;
    const int rV = tid >> 8;   // 0..1
    const int dV = tid & 255;

    float hk[32], hv[32];
    #pragma unroll
    for (int i = 0; i < 32; ++i) { hk[i] = 0.f; hv[i] = 0.f; }

    __shared__ float qs[256], ks[256], vs[256];
    __shared__ float decs[MSLOT], ss[MSLOT], qvs[MSLOT];
    __shared__ float p1s[512], pOs[512], red[4];

    const int qoff = h * DKH;
    const int koff = DMODEL + h * DKH;
    const int voff = 2 * DMODEL + h * DKH;
    const int goff = 3 * DMODEL + h * MSLOT;
    const size_t base = (size_t)b * T_SEQ * YCOLS;

    for (int t = 0; t < T_SEQ; ++t) {
        const u16* Yr = Y + base + (size_t)t * YCOLS;
        if (tid < 256) {
            qs[tid] = bf2f(Yr[qoff + tid]);
            vs[tid] = bf2f(Yr[voff + tid]);
        } else {
            int i = tid - 256;
            ks[i] = bf2f(Yr[koff + i]);
            if (i < MSLOT) {
                float g = bf2f(Yr[goff + i]);     // log-forget gate
                float d = __expf(g);              // decay
                decs[i] = d;
                ss[i]   = 1.f - d;                // write strength s = 1 - exp(g)
            }
        }
        __syncthreads();

        // --- pass 1: Hk update + partial o1 ---
        {
            const float dec = decs[mK], s = ss[mK];
            float p1 = 0.f;
            #pragma unroll
            for (int i = 0; i < 32; ++i) {
                int d = cK * 32 + i;
                hk[i] = hk[i] * dec + ks[d] * s;
                p1 += qs[d] * hk[i];
            }
            p1s[cK * 64 + mK] = p1;
        }
        __syncthreads();

        // --- softmax over 64 slots (threads 0..63 = 2 full waves) ---
        float o1 = 0.f, e = 0.f;
        if (tid < 64) {
            #pragma unroll
            for (int c2 = 0; c2 < 8; ++c2) o1 += p1s[c2 * 64 + tid];
            float mx = o1;
            #pragma unroll
            for (int off = 16; off; off >>= 1) mx = fmaxf(mx, __shfl_xor(mx, off, 32));
            if ((tid & 31) == 0) red[tid >> 5] = mx;
        }
        __syncthreads();
        if (tid < 64) {
            float gm = fmaxf(red[0], red[1]);
            e = __expf(o1 - gm);
            float sm = e;
            #pragma unroll
            for (int off = 16; off; off >>= 1) sm += __shfl_xor(sm, off, 32);
            if ((tid & 31) == 0) red[2 + (tid >> 5)] = sm;
        }
        __syncthreads();
        if (tid < 64) qvs[tid] = e / (red[2] + red[3]);
        __syncthreads();

        // --- pass 2: Hv update + output ---
        {
            const float vv = vs[dV];
            float po = 0.f;
            #pragma unroll
            for (int i = 0; i < 32; ++i) {
                int m = rV * 32 + i;
                hv[i] = hv[i] * decs[m] + ss[m] * vv;
                po += qvs[m] * hv[i];
            }
            pOs[rV * 256 + dV] = po;
        }
        __syncthreads();
        if (tid < 256) {
            O[((size_t)(b * T_SEQ + t)) * DMODEL + h * DKH + tid] =
                pOs[tid] + pOs[256 + tid];
        }
        // next-iter LDS writes are ordered against this iter's reads by the
        // barrier above (all pass-2 reads precede each thread's pOs store).
    }
}

// ---------------------------------------------------------------------------
// Orchestration
// ---------------------------------------------------------------------------
static inline size_t align_up(size_t x) { return (x + 255) & ~(size_t)255; }

extern "C" void kernel_launch(void* const* d_in, const int* in_sizes, int n_in,
                              void* d_out, int out_size, void* d_ws, size_t ws_size,
                              hipStream_t stream) {
    (void)in_sizes; (void)n_in; (void)out_size; (void)ws_size;
    const float* hid      = (const float*)d_in[0];
    const float* norm_w   = (const float*)d_in[1];
    const float* q_w      = (const float*)d_in[2];
    const float* k_w      = (const float*)d_in[3];
    const float* v_w      = (const float*)d_in[4];
    const float* f_w      = (const float*)d_in[5];
    const float* g_norm_w = (const float*)d_in[6];
    const float* o_w      = (const float*)d_in[7];
    float* out = (float*)d_out;

    char* ws = (char*)d_ws;
    size_t off = 0;
    auto take = [&](size_t bytes) -> char* {
        char* p = ws + off;
        off += align_up(bytes);
        return p;
    };
    const size_t NTOK = (size_t)BATCH * T_SEQ;                 // 8192
    u16*   Wcat = (u16*)take((size_t)YCOLS * KDIM * 2);        // q,k,v,f rows
    u16*   Wo   = (u16*)take((size_t)DMODEL * KDIM * 2);
    u16*   Xn   = (u16*)take(NTOK * DMODEL * 2);
    u16*   Yq   = (u16*)take(NTOK * YCOLS * 2);
    float* Os   = (float*)take(NTOK * DMODEL * 4);
    u16*   On   = (u16*)take(NTOK * DMODEL * 2);

    // 1) weights -> bf16 (concat layout: q|k|v|f row blocks)
    const int ndd = DMODEL * KDIM;             // 4.19M
    const int nf  = NHEAD * MSLOT * KDIM;      // 1.05M
    cvt_kernel<<<ndd / 1024, 256, 0, stream>>>(q_w, Wcat + (size_t)0 * ndd, ndd);
    cvt_kernel<<<ndd / 1024, 256, 0, stream>>>(k_w, Wcat + (size_t)1 * ndd, ndd);
    cvt_kernel<<<ndd / 1024, 256, 0, stream>>>(v_w, Wcat + (size_t)2 * ndd, ndd);
    cvt_kernel<<<nf  / 1024, 256, 0, stream>>>(f_w, Wcat + (size_t)3 * ndd, nf);
    cvt_kernel<<<ndd / 1024, 256, 0, stream>>>(o_w, Wo, ndd);

    // 2) rmsnorm(x) -> bf16
    rmsnorm_cast_kernel<<<(int)NTOK, 256, 0, stream>>>(hid, norm_w, Xn, 0);

    // 3) fused q/k/v/f projection with activation epilogue
    wmma_gemm_kernel<<<dim3(NTOK / 128, YCOLS / 256), 256, 0, stream>>>(
        Xn, Wcat, (void*)Yq, YCOLS, KDIM, 1);

    // 4) sequential gated-slot scan (one WG per (b,h))
    gsa_scan_kernel<<<BATCH * NHEAD, 512, 0, stream>>>(Yq, Os);

    // 5) rmsnorm(swish(o)) -> bf16
    rmsnorm_cast_kernel<<<(int)NTOK, 256, 0, stream>>>(Os, g_norm_w, On, 1);

    // 6) output projection -> fp32
    wmma_gemm_kernel<<<dim3(NTOK / 128, DMODEL / 256), 256, 0, stream>>>(
        On, Wo, (void*)out, DMODEL, KDIM, 0);
}
